// BiLSTMCRF_80255758893452
// MI455X (gfx1250) — compile-verified, matrix operand order checked by reading, not run on hardware
//
#include <hip/hip_runtime.h>
#include <hip/hip_bf16.h>
#include <math.h>

// ---------------------------------------------------------------------------
// BiLSTM-CRF for MI455X (gfx1250, wave32, WMMA bf16 16x16x32, f32 accum)
// ---------------------------------------------------------------------------

#define Bc   32
#define Sc   512
#define Ec   512
#define HIDc 1024
#define HDc  512
#define Gc   2048   // 4*HD
#define Tc   8

typedef __attribute__((ext_vector_type(16))) __bf16 v16bf;
typedef __attribute__((ext_vector_type(8)))  float  v8f;

union Frag16 { v16bf v; unsigned u[8]; };

static __device__ __forceinline__ __bf16 f2bf(float f) {
  union { float f; unsigned u; } in; in.f = f;
  unsigned r = in.u + 0x7FFFu + ((in.u >> 16) & 1u);   // round-to-nearest-even
  union { unsigned short s; __bf16 b; } out; out.s = (unsigned short)(r >> 16);
  return out.b;
}

static __device__ __forceinline__ float sigf(float x) {
  return 1.0f / (1.0f + __expf(-x));
}

// ---------------------------------------------------------------------------
// f32 -> bf16 bulk convert
// ---------------------------------------------------------------------------
__global__ void cvt_f32_bf16(const float* __restrict__ src,
                             __bf16* __restrict__ dst, long n) {
  long i = (long)blockIdx.x * blockDim.x + threadIdx.x;
  if (i < n) dst[i] = f2bf(src[i]);
}

// ---------------------------------------------------------------------------
// embedding gather -> bf16  (x0_bf laid out (B,S,E) row-major)
// ---------------------------------------------------------------------------
__global__ void embed_bf16(const float* __restrict__ emb,
                           const int* __restrict__ chars,
                           __bf16* __restrict__ xbf, long total_rows, int E) {
  long i = (long)blockIdx.x * blockDim.x + threadIdx.x;
  if (i >= total_rows * E) return;
  long r = i / E; int e = (int)(i - r * E);
  xbf[i] = f2bf(emb[(size_t)chars[r] * E + e]);
}

// ---------------------------------------------------------------------------
// WMMA GEMM: Y[m,n] = sum_k A[src(m),k] * W[n,k] + bias[n]
//   logical row m = s*B + b  (matches xp layout (S,B,G))
//   A source row = (b, s) or (b, rev(s, L[b])) of X laid out (B,S,K)
//   One wave per 16x16 output tile, 8 waves / block.
// ---------------------------------------------------------------------------
__global__ __launch_bounds__(256)
void wmma_gemm_bias(const __bf16* __restrict__ X,
                    const __bf16* __restrict__ Wt,   // (N,K) row-major
                    const float*  __restrict__ bias, // (N)
                    float* __restrict__ Y,           // (M,N) row-major
                    const int* __restrict__ seq_len,
                    int Sdim, int Bdim, int N, int K, int reverse) {
  const int wv   = threadIdx.x >> 5;
  const int lane = threadIdx.x & 31;
  const int nl   = lane & 15;
  const int half8 = (lane >> 4) * 8;

  const int  Nt = N >> 4;
  const long Mt = ((long)Sdim * Bdim) >> 4;
  long t = (long)blockIdx.x * 8 + wv;
  if (t >= Mt * (long)Nt) return;                 // wave-uniform exit
  const int  nt = (int)(t % Nt);
  const long mt = t / Nt;

  // per-lane A row (each lane owns one row of the 16-row tile)
  const int m = (int)(mt * 16) + nl;              // logical row, s-major
  const int b = m % Bdim;
  const int s = m / Bdim;
  int src_s = s;
  if (reverse) { int L = seq_len[b]; src_s = (s < L) ? (L - 1 - s) : s; }
  const __bf16* pA = X  + ((size_t)b * Sdim + src_s) * K;
  const __bf16* pB = Wt + (size_t)(nt * 16 + nl) * K;

  v8f acc;
  const float bv = bias[nt * 16 + nl];
#pragma unroll
  for (int r = 0; r < 8; ++r) acc[r] = bv;

  const int KT = K >> 5;
  for (int kt = 0; kt < KT; ++kt) {
    const int kb = kt * 32;
    Frag16 a, w;
#pragma unroll
    for (int r = 0; r < 4; ++r) {
      a.u[r]     = *(const unsigned*)(pA + kb + 2 * r + half8);
      a.u[4 + r] = *(const unsigned*)(pA + kb + 16 + 2 * r + half8);
      w.u[r]     = *(const unsigned*)(pB + kb + 2 * r + half8);
      w.u[4 + r] = *(const unsigned*)(pB + kb + 16 + 2 * r + half8);
    }
    acc = __builtin_amdgcn_wmma_f32_16x16x32_bf16(false, a.v, false, w.v,
                                                  (short)0, acc, false, false);
  }

  // C layout: VGPR r -> row r + 8*(lane/16), col = lane&15
  float* pC = Y + (size_t)(mt * 16) * N + (size_t)nt * 16 + nl;
  const int halfm = (lane >> 4) * 8;
#pragma unroll
  for (int r = 0; r < 8; ++r)
    pC[(size_t)(r + halfm) * N] = acc[r];
}

// ---------------------------------------------------------------------------
// LSTM scan: one workgroup (32 waves) per direction.
//   gates = xp[s] + h @ w_hh.T  via WMMA; c in registers, h in LDS (double buf)
//   Wave w owns columns j in [16w,16w+16): i/f/g/o for the same (b,j) land
//   in the same lane/VGPR slot -> elementwise update stays in registers.
//   Output written masked + time-unreversed straight into hcat / xnext_bf.
// ---------------------------------------------------------------------------
__global__ __launch_bounds__(1024)
void lstm_scan(const float* __restrict__ xp_f,
               const float* __restrict__ xp_b,
               const __bf16* __restrict__ whh_bf,   // (2, G, HD) bf16
               const int* __restrict__ seq_len,
               float* __restrict__ hcat,            // (B, S, HID) f32
               __bf16* __restrict__ xnext_bf,       // (B, S, HID) bf16
               int S) {
  __shared__ __bf16 hbuf[2][Bc * HDc];              // 2 x 32KB double buffer

  const int dir = blockIdx.x;
  const float*  xp = dir ? xp_b : xp_f;
  const __bf16* W  = whh_bf + (size_t)dir * Gc * HDc;

  const int tid  = threadIdx.x;
  const int wv   = tid >> 5;
  const int lane = tid & 31;
  const int nl   = lane & 15;
  const int half  = lane >> 4;
  const int half8 = half * 8;
  const int jbase = wv * 16;
  const int j     = jbase + nl;

  for (int i = tid; i < Bc * HDc; i += 1024) hbuf[0][i] = f2bf(0.0f);

  float c_reg[2][8];
  int   Lr[2][8];
#pragma unroll
  for (int mt = 0; mt < 2; ++mt)
#pragma unroll
    for (int r = 0; r < 8; ++r) {
      c_reg[mt][r] = 0.0f;
      Lr[mt][r] = seq_len[mt * 16 + r + 8 * half];
    }
  __syncthreads();

  for (int s = 0; s < S; ++s) {
    const int cur = s & 1, nxt = cur ^ 1;
    const float* xps = xp + (size_t)s * (Bc * Gc);

    if (s + 1 < S)   // warm L2 with next step's gate preactivations
      __builtin_prefetch(xp + (size_t)(s + 1) * (Bc * Gc) + tid * 16, 0, 1);

#pragma unroll
    for (int mt = 0; mt < 2; ++mt) {
      v8f ai, af, ag, ao;
#pragma unroll
      for (int r = 0; r < 8; ++r) {
        const float* row = xps + (size_t)(mt * 16 + r + 8 * half) * Gc;
        ai[r] = row[0 * HDc + j];
        af[r] = row[1 * HDc + j];
        ag[r] = row[2 * HDc + j];
        ao[r] = row[3 * HDc + j];
      }

      const __bf16* hrow = &hbuf[cur][(mt * 16 + nl) * HDc];
#pragma unroll
      for (int kt = 0; kt < HDc / 32; ++kt) {
        const int kb = kt * 32;
        Frag16 a;
#pragma unroll
        for (int r = 0; r < 4; ++r) {
          a.u[r]     = *(const unsigned*)(hrow + kb + 2 * r + half8);
          a.u[4 + r] = *(const unsigned*)(hrow + kb + 16 + 2 * r + half8);
        }
        const __bf16* w0 = W + (size_t)(0 * HDc + j) * HDc + kb;
        const __bf16* w1 = W + (size_t)(1 * HDc + j) * HDc + kb;
        const __bf16* w2 = W + (size_t)(2 * HDc + j) * HDc + kb;
        const __bf16* w3 = W + (size_t)(3 * HDc + j) * HDc + kb;
        Frag16 bi, bf, bg, bo;
#pragma unroll
        for (int r = 0; r < 4; ++r) {
          bi.u[r]     = *(const unsigned*)(w0 + 2 * r + half8);
          bi.u[4 + r] = *(const unsigned*)(w0 + 16 + 2 * r + half8);
          bf.u[r]     = *(const unsigned*)(w1 + 2 * r + half8);
          bf.u[4 + r] = *(const unsigned*)(w1 + 16 + 2 * r + half8);
          bg.u[r]     = *(const unsigned*)(w2 + 2 * r + half8);
          bg.u[4 + r] = *(const unsigned*)(w2 + 16 + 2 * r + half8);
          bo.u[r]     = *(const unsigned*)(w3 + 2 * r + half8);
          bo.u[4 + r] = *(const unsigned*)(w3 + 16 + 2 * r + half8);
        }
        ai = __builtin_amdgcn_wmma_f32_16x16x32_bf16(false, a.v, false, bi.v, (short)0, ai, false, false);
        af = __builtin_amdgcn_wmma_f32_16x16x32_bf16(false, a.v, false, bf.v, (short)0, af, false, false);
        ag = __builtin_amdgcn_wmma_f32_16x16x32_bf16(false, a.v, false, bg.v, (short)0, ag, false, false);
        ao = __builtin_amdgcn_wmma_f32_16x16x32_bf16(false, a.v, false, bo.v, (short)0, ao, false, false);
      }

      // LSTM cell update entirely in registers
#pragma unroll
      for (int r = 0; r < 8; ++r) {
        const float iv = sigf(ai[r]);
        const float fv = sigf(af[r]);
        const float gv = tanhf(ag[r]);
        const float ov = sigf(ao[r]);
        const float c  = fv * c_reg[mt][r] + iv * gv;
        c_reg[mt][r] = c;
        const float h = ov * tanhf(c);

        const int b = mt * 16 + r + 8 * half;
        hbuf[nxt][b * HDc + j] = f2bf(h);

        const int L  = Lr[mt][r];
        const int sd = dir ? ((s < L) ? (L - 1 - s) : s) : s;
        const float hv = (sd < L) ? h : 0.0f;
        const size_t o = ((size_t)b * S + sd) * HIDc + dir * HDc + j;
        hcat[o]     = hv;
        xnext_bf[o] = f2bf(hv);
      }
    }
    __syncthreads();
  }
}

// ---------------------------------------------------------------------------
// FC (HID -> T=8) + log_softmax, one thread per (b,s)
// ---------------------------------------------------------------------------
__global__ void fc_logsoftmax(const float* __restrict__ hcat,
                              const float* __restrict__ fc_w,  // (T,HID)
                              const float* __restrict__ fc_b,
                              float* __restrict__ logits,      // (B,S,T)
                              int total) {
  int m = blockIdx.x * blockDim.x + threadIdx.x;
  if (m >= total) return;
  const float* x = hcat + (size_t)m * HIDc;
  float z[Tc];
#pragma unroll
  for (int t = 0; t < Tc; ++t) z[t] = fc_b[t];
  for (int k = 0; k < HIDc; ++k) {
    const float xv = x[k];
#pragma unroll
    for (int t = 0; t < Tc; ++t) z[t] += xv * fc_w[t * HIDc + k];
  }
  float mx = z[0];
#pragma unroll
  for (int t = 1; t < Tc; ++t) mx = fmaxf(mx, z[t]);
  float sum = 0.0f;
#pragma unroll
  for (int t = 0; t < Tc; ++t) sum += __expf(z[t] - mx);
  const float ls = mx + __logf(sum);
#pragma unroll
  for (int t = 0; t < Tc; ++t) logits[(size_t)m * Tc + t] = z[t] - ls;
}

// ---------------------------------------------------------------------------
// CRF negative log-likelihood, one thread per batch element (1 wave total)
// ---------------------------------------------------------------------------
__global__ void crf_nll(const float* __restrict__ logits,
                        const int* __restrict__ target,
                        const int* __restrict__ seq_len,
                        const float* __restrict__ trans,
                        float* __restrict__ out, int S) {
  __shared__ float tr[Tc * Tc];
  const int b = threadIdx.x;
  tr[b] = trans[b]; tr[b + 32] = trans[b + 32];
  __syncthreads();

  const float* lg = logits + (size_t)b * S * Tc;
  const int*   tg = target + (size_t)b * S;
  const int L = seq_len[b];

  float alpha[Tc];
#pragma unroll
  for (int t = 0; t < Tc; ++t) alpha[t] = lg[t];

  int   pt   = tg[0];
  float gold = (0 < L) ? lg[pt] : 0.0f;

  for (int s = 1; s < S; ++s) {
    const float* l = lg + (size_t)s * Tc;
    const int t2 = tg[s];
    if (s < L) {
      float na[Tc];
#pragma unroll
      for (int jn = 0; jn < Tc; ++jn) {
        float mx = alpha[0] + tr[jn];
#pragma unroll
        for (int i = 1; i < Tc; ++i) mx = fmaxf(mx, alpha[i] + tr[i * Tc + jn]);
        float sm = 0.0f;
#pragma unroll
        for (int i = 0; i < Tc; ++i) sm += __expf(alpha[i] + tr[i * Tc + jn] - mx);
        na[jn] = mx + __logf(sm) + l[jn];
      }
#pragma unroll
      for (int jn = 0; jn < Tc; ++jn) alpha[jn] = na[jn];
      gold += l[t2] + tr[pt * Tc + t2];   // mask is monotone: s<L => s-1<L
    }
    pt = t2;
  }
  float mx = alpha[0];
#pragma unroll
  for (int t = 1; t < Tc; ++t) mx = fmaxf(mx, alpha[t]);
  float sm = 0.0f;
#pragma unroll
  for (int t = 0; t < Tc; ++t) sm += __expf(alpha[t] - mx);
  out[b] = (mx + __logf(sm)) - gold;
}

// ---------------------------------------------------------------------------
// Host-side orchestration
// ---------------------------------------------------------------------------
extern "C" void kernel_launch(void* const* d_in, const int* in_sizes, int n_in,
                              void* d_out, int out_size, void* d_ws, size_t ws_size,
                              hipStream_t stream) {
  (void)in_sizes; (void)n_in; (void)out_size; (void)ws_size;
  const int*   chars   = (const int*)  d_in[0];
  const int*   seq_len = (const int*)  d_in[1];
  const int*   target  = (const int*)  d_in[2];
  const float* emb     = (const float*)d_in[3];
  const float* w_ih0   = (const float*)d_in[4];
  const float* w_hh0   = (const float*)d_in[5];
  const float* b0      = (const float*)d_in[6];
  const float* w_ih1   = (const float*)d_in[7];
  const float* w_hh1   = (const float*)d_in[8];
  const float* b1      = (const float*)d_in[9];
  const float* fc_w    = (const float*)d_in[10];
  const float* fc_b    = (const float*)d_in[11];
  const float* trans   = (const float*)d_in[12];
  float* out = (float*)d_out;

  char* ws = (char*)d_ws;
  size_t off = 0;
  auto get = [&](size_t bytes) -> void* {
    void* p = ws + off;
    off = (off + bytes + 255) & ~(size_t)255;
    return p;
  };
  __bf16* wih0_bf = (__bf16*)get((size_t)2 * Gc * Ec   * 2);
  __bf16* whh0_bf = (__bf16*)get((size_t)2 * Gc * HDc  * 2);
  __bf16* wih1_bf = (__bf16*)get((size_t)2 * Gc * HIDc * 2);
  __bf16* whh1_bf = (__bf16*)get((size_t)2 * Gc * HDc  * 2);
  __bf16* x0_bf   = (__bf16*)get((size_t)Bc * Sc * Ec   * 2);
  __bf16* x1_bf   = (__bf16*)get((size_t)Bc * Sc * HIDc * 2);
  float*  xp_f    = (float*) get((size_t)Sc * Bc * Gc * 4);
  float*  xp_b    = (float*) get((size_t)Sc * Bc * Gc * 4);
  float*  hcat    = (float*) get((size_t)Bc * Sc * HIDc * 4);
  float*  logits  = (float*) get((size_t)Bc * Sc * Tc * 4);

  // weights -> bf16 (L2-resident afterwards)
  {
    long n;
    n = (long)2 * Gc * Ec;
    cvt_f32_bf16<<<(int)((n + 255) / 256), 256, 0, stream>>>(w_ih0, wih0_bf, n);
    n = (long)2 * Gc * HDc;
    cvt_f32_bf16<<<(int)((n + 255) / 256), 256, 0, stream>>>(w_hh0, whh0_bf, n);
    n = (long)2 * Gc * HIDc;
    cvt_f32_bf16<<<(int)((n + 255) / 256), 256, 0, stream>>>(w_ih1, wih1_bf, n);
    n = (long)2 * Gc * HDc;
    cvt_f32_bf16<<<(int)((n + 255) / 256), 256, 0, stream>>>(w_hh1, whh1_bf, n);
  }

  // embedding gather
  {
    long n = (long)Bc * Sc * Ec;
    embed_bf16<<<(int)((n + 255) / 256), 256, 0, stream>>>(emb, chars, x0_bf,
                                                           (long)Bc * Sc, Ec);
  }

  const int tiles  = (Sc * Bc / 16) * (Gc / 16);  // 131072
  const int gblks  = tiles / 8;                   // 16384

  // ---- layer 0 ----
  wmma_gemm_bias<<<gblks, 256, 0, stream>>>(x0_bf, wih0_bf, b0, xp_f,
                                            seq_len, Sc, Bc, Gc, Ec, 0);
  wmma_gemm_bias<<<gblks, 256, 0, stream>>>(x0_bf, wih0_bf + (size_t)Gc * Ec,
                                            b0 + Gc, xp_b,
                                            seq_len, Sc, Bc, Gc, Ec, 1);
  lstm_scan<<<2, 1024, 0, stream>>>(xp_f, xp_b, whh0_bf, seq_len,
                                    hcat, x1_bf, Sc);

  // ---- layer 1 ----
  wmma_gemm_bias<<<gblks, 256, 0, stream>>>(x1_bf, wih1_bf, b1, xp_f,
                                            seq_len, Sc, Bc, Gc, HIDc, 0);
  wmma_gemm_bias<<<gblks, 256, 0, stream>>>(x1_bf, wih1_bf + (size_t)Gc * HIDc,
                                            b1 + Gc, xp_b,
                                            seq_len, Sc, Bc, Gc, HIDc, 1);
  lstm_scan<<<2, 1024, 0, stream>>>(xp_f, xp_b, whh1_bf, seq_len,
                                    hcat, x1_bf, Sc);

  // ---- head ----
  fc_logsoftmax<<<(Bc * Sc + 255) / 256, 256, 0, stream>>>(hcat, fc_w, fc_b,
                                                           logits, Bc * Sc);
  crf_nll<<<1, 32, 0, stream>>>(logits, target, seq_len, trans, out, Sc);
}
